// GCN_61632780698356
// MI455X (gfx1250) — compile-verified, matrix operand order checked by reading, not run on hardware
//
#include <hip/hip_runtime.h>
#include <hip/hip_bf16.h>

#define NN 50000
#define NE 800000

typedef __attribute__((ext_vector_type(16))) __bf16 v16bf;
typedef __attribute__((ext_vector_type(8)))  __bf16 v8bf;
typedef __attribute__((ext_vector_type(8)))  float  v8f;

// ---- bf16 helpers (bit-exact RNE via integer ops) ----
static __device__ __forceinline__ __bf16 f2bf(float x) {
    unsigned u = __float_as_uint(x);
    unsigned r = (u + 0x7fffu + ((u >> 16) & 1u)) >> 16;   // round-to-nearest-even
    unsigned short s = (unsigned short)r;
    return __builtin_bit_cast(__bf16, s);
}
static __device__ __forceinline__ float bf2f(__bf16 h) {
    unsigned short s = __builtin_bit_cast(unsigned short, h);
    return __uint_as_float(((unsigned)s) << 16);
}

static __device__ __forceinline__ v8f wmma_bf16(v16bf a, v16bf b, v8f c) {
    return __builtin_amdgcn_wmma_f32_16x16x32_bf16(false, a, false, b, (short)0, c,
                                                   false, false);
}

static __device__ __forceinline__ v16bf load_frag(const __bf16* p) {
    v8bf lo8 = *(const v8bf*)p;
    v8bf hi8 = *(const v8bf*)(p + 16);
    return __builtin_shufflevector(lo8, hi8, 0,1,2,3,4,5,6,7,8,9,10,11,12,13,14,15);
}

// ---- gfx1250 async global->LDS copy (16B per lane), ASYNCcnt-tracked ----
static __device__ __forceinline__ void async_copy16(unsigned lds_off, const void* gptr) {
    asm volatile("global_load_async_to_lds_b128 %0, %1, off"
                 :: "v"(lds_off), "v"(gptr) : "memory");
}
static __device__ __forceinline__ void wait_async0() {
    asm volatile("s_wait_asynccnt 0x0" ::: "memory");
}

// ---- degree / normalization ----
__global__ void k_zero_u32(unsigned* __restrict__ p, int n) {
    int i = blockIdx.x * blockDim.x + threadIdx.x;
    if (i < n) p[i] = 0u;
}
__global__ void k_count(const long long* __restrict__ dst, unsigned* __restrict__ deg, int e) {
    int i = blockIdx.x * blockDim.x + threadIdx.x;
    if (i < e) atomicAdd(&deg[(int)dst[i]], 1u);
}
__global__ void k_dinv(const unsigned* __restrict__ deg, float* __restrict__ dinv, int n) {
    int i = blockIdx.x * blockDim.x + threadIdx.x;
    if (i < n) dinv[i] = rsqrtf((float)deg[i] + 1.0f);
}

// ---- split fp32 array -> bf16 hi/lo (activations) ----
__global__ void k_split(const float* __restrict__ X, __bf16* __restrict__ hi,
                        __bf16* __restrict__ lo, int total) {
    int i = blockIdx.x * blockDim.x + threadIdx.x;
    if (i >= total) return;
    float x = X[i];
    __bf16 h = f2bf(x);
    hi[i] = h;
    lo[i] = f2bf(x - bf2f(h));
}

// ---- weight prep: fp32 [K,M] (transpose=1) or [M,K] (transpose=0) -> bf16 hi/lo [Mpad,K] ----
__global__ void k_prep(const float* __restrict__ W, __bf16* __restrict__ bhi,
                       __bf16* __restrict__ blo, int K, int M, int Mpad, int transpose) {
    int i = blockIdx.x * blockDim.x + threadIdx.x;
    int total = Mpad * K;
    if (i >= total) return;
    int m = i / K, k = i - m * K;
    float x = 0.0f;
    if (m < M) x = transpose ? W[(size_t)k * M + m] : W[(size_t)m * K + k];
    __bf16 h = f2bf(x);
    bhi[i] = h;
    blo[i] = f2bf(x - bf2f(h));
}

// ---- WMMA GEMM: C[N,M] = A[N,K] @ Bt[Mpad,K]^T (+bias)(+relu) ----
// Block = 4 waves sharing one 64-col strip; B strip staged in LDS via async copies.
// A fragments double-buffered; 12 WMMAs interleaved across 4 accumulators.
__global__ void k_gemm(const __bf16* __restrict__ Ahi, const __bf16* __restrict__ Alo,
                       const __bf16* __restrict__ Bhi, const __bf16* __restrict__ Blo,
                       float* __restrict__ C, const float* __restrict__ bias,
                       int K, int M, int relu, int strips, int rowTiles) {
    extern __shared__ __bf16 sh[];
    const int pitch = K + 8;                      // dword-bank skew: stride % 64 banks != 0
    int strip  = blockIdx.x % strips;
    int rowBlk = blockIdx.x / strips;
    int widx   = threadIdx.x >> 5;
    int lane   = threadIdx.x & 31;
    int rowTile = rowBlk * 4 + widx;
    if (rowTile >= rowTiles) rowTile = rowTiles - 1;   // clamp: all waves reach barrier

    // ---- stage B strip (hi then lo) into LDS with async global->LDS copies ----
    {
        unsigned shbase = (unsigned)(uintptr_t)(void*)sh;  // low 32 bits = LDS offset
        int chunksPerRow = K / 8;                          // 16-byte chunks per row
        int totalChunks  = 64 * chunksPerRow;
        const __bf16* gb0 = Bhi + (size_t)strip * 64 * K;
        const __bf16* gb1 = Blo + (size_t)strip * 64 * K;
        for (int ch = (int)threadIdx.x; ch < totalChunks; ch += 128) {
            int c = ch / chunksPerRow, kk = (ch - c * chunksPerRow) * 8;
            unsigned loff = shbase + (unsigned)(c * pitch + kk) * 2u;
            async_copy16(loff, (const void*)(gb0 + (size_t)c * K + kk));
            async_copy16(loff + (unsigned)(64 * pitch) * 2u,
                         (const void*)(gb1 + (size_t)c * K + kk));
        }
        wait_async0();
        __syncthreads();
    }

    int mrow = rowTile * 16 + (lane & 15);
    int kc   = (lane >> 4) * 8;
    const __bf16* arh  = Ahi + (size_t)mrow * K + kc;
    const __bf16* arl  = Alo + (size_t)mrow * K + kc;
    const __bf16* shlo = sh + 64 * pitch;
    int cbase[4];
#pragma unroll
    for (int j = 0; j < 4; ++j)
        cbase[j] = (j * 16 + (lane & 15)) * pitch + kc;

    v8f acc[4] = {v8f{}, v8f{}, v8f{}, v8f{}};
    v16bf ahi = load_frag(arh);
    v16bf alo = load_frag(arl);

    for (int k0 = 0; k0 < K; k0 += 32) {
        int kn = (k0 + 32 < K) ? k0 + 32 : 0;          // unconditional prefetch
        v16bf ahiN = load_frag(arh + kn);
        v16bf aloN = load_frag(arl + kn);
        v16bf bhi[4], blo[4];
#pragma unroll
        for (int j = 0; j < 4; ++j) bhi[j] = load_frag(sh   + cbase[j] + k0);
#pragma unroll
        for (int j = 0; j < 4; ++j) blo[j] = load_frag(shlo + cbase[j] + k0);
#pragma unroll
        for (int j = 0; j < 4; ++j) acc[j] = wmma_bf16(ahi, bhi[j], acc[j]);
#pragma unroll
        for (int j = 0; j < 4; ++j) acc[j] = wmma_bf16(alo, bhi[j], acc[j]);
#pragma unroll
        for (int j = 0; j < 4; ++j) acc[j] = wmma_bf16(ahi, blo[j], acc[j]);
        ahi = ahiN;
        alo = aloN;
    }

    // store per ISA C/D layout: VGPR v -> (M = v or v+8, N = lane&15)
#pragma unroll
    for (int j = 0; j < 4; ++j) {
        int col = strip * 64 + j * 16 + (lane & 15);
        if (col < M) {
            float bv = bias ? bias[col] : 0.0f;
#pragma unroll
            for (int v = 0; v < 8; ++v) {
                int row   = rowTile * 16 + ((lane & 16) ? v + 8 : v);
                float val = acc[j][v] + bv;
                if (relu) val = fmaxf(val, 0.0f);
                C[(size_t)row * M + col] = val;
            }
        }
    }
}

// ---- self-loop + bias init: out = xw * dinv^2 + b ----
__global__ void k_selfinit(const float* __restrict__ xw, const float* __restrict__ dinv,
                           const float* __restrict__ b, float* __restrict__ out,
                           int D, int total) {
    int i = blockIdx.x * blockDim.x + threadIdx.x;
    if (i >= total) return;
    int row = i / D;
    int d   = i - row * D;
    float di = dinv[row];
    out[i] = xw[i] * di * di + b[d];
}

// ---- edge scatter-add: out[dst] += xw[src] * dinv[src]*dinv[dst] (one wave per edge) ----
__global__ void k_edge(const long long* __restrict__ src, const long long* __restrict__ dst,
                       const float* __restrict__ dinv, const float* __restrict__ xw,
                       float* __restrict__ out, int D, int E) {
    int wave = blockIdx.x * (blockDim.x >> 5) + (threadIdx.x >> 5);
    if (wave >= E) return;
    int lane = threadIdx.x & 31;
    int s = (int)src[wave], d = (int)dst[wave];
    float c = dinv[s] * dinv[d];
    const float* xr = xw + (size_t)s * D;
    float* orow = out + (size_t)d * D;
    for (int i = lane * 4; i < D; i += 128) {
        float4 v = *(const float4*)(xr + i);
        unsafeAtomicAdd(orow + i + 0, v.x * c);
        unsafeAtomicAdd(orow + i + 1, v.y * c);
        unsafeAtomicAdd(orow + i + 2, v.z * c);
        unsafeAtomicAdd(orow + i + 3, v.w * c);
    }
}

extern "C" void kernel_launch(void* const* d_in, const int* in_sizes, int n_in,
                              void* d_out, int out_size, void* d_ws, size_t ws_size,
                              hipStream_t stream) {
    const float*     nodes = (const float*)d_in[0];
    const long long* edges = (const long long*)d_in[1];   // int64 [2, E]
    const float*     label = (const float*)d_in[2];       // [1000, 128]
    const float* W1 = (const float*)d_in[3];  const float* b1 = (const float*)d_in[4];
    const float* W2 = (const float*)d_in[5];  const float* b2 = (const float*)d_in[6];
    const float* W3 = (const float*)d_in[7];  const float* b3 = (const float*)d_in[8];
    const float* fc1w = (const float*)d_in[9];  const float* fc1b = (const float*)d_in[10];
    const float* fc2w = (const float*)d_in[11]; const float* fc2b = (const float*)d_in[12];
    float* out = (float*)d_out;

    const long long* esrc = edges;
    const long long* edst = edges + NE;

    // workspace carve-up
    char* ws = (char*)d_ws;
    unsigned* deg  = (unsigned*)ws;           ws += (size_t)NN * sizeof(unsigned);
    float*    dinv = (float*)ws;              ws += (size_t)NN * sizeof(float);
    float*    bufA = (float*)ws;              ws += (size_t)NN * 256 * sizeof(float);
    float*    bufB = (float*)ws;              ws += (size_t)NN * 256 * sizeof(float);
    __bf16*   Ahi  = (__bf16*)ws;             ws += (size_t)NN * 512 * sizeof(__bf16);
    __bf16*   Alo  = (__bf16*)ws;             ws += (size_t)NN * 512 * sizeof(__bf16);
    __bf16*   Bhi  = (__bf16*)ws;             ws += (size_t)131072 * sizeof(__bf16);
    __bf16*   Blo  = (__bf16*)ws;

    const int rowTiles = NN / 16;   // 3125

    auto prep = [&](const float* W, int K, int M, int Mpad, int transpose) {
        int total = Mpad * K;
        k_prep<<<(total + 255) / 256, 256, 0, stream>>>(W, Bhi, Blo, K, M, Mpad, transpose);
    };
    auto gemm = [&](const float* A, float* C, const float* bias, int K, int M, int Mpad,
                    int relu) {
        int totalA = NN * K;
        k_split<<<(totalA + 255) / 256, 256, 0, stream>>>(A, Ahi, Alo, totalA);
        int strips  = Mpad / 64;
        int blocks  = strips * ((rowTiles + 3) / 4);
        size_t shmem = (size_t)(2 * 64 * (K + 8)) * sizeof(__bf16);
        k_gemm<<<blocks, 128, shmem, stream>>>(Ahi, Alo, Bhi, Blo, C, bias, K, M,
                                               relu, strips, rowTiles);
    };
    auto gcn_agg = [&](const float* xw, const float* b, float* outbuf, int D) {
        k_selfinit<<<(NN * D + 255) / 256, 256, 0, stream>>>(xw, dinv, b, outbuf, D, NN * D);
        k_edge<<<(NE + 7) / 8, 256, 0, stream>>>(esrc, edst, dinv, xw, outbuf, D, NE);
    };

    // degree / dinv
    k_zero_u32<<<(NN + 255) / 256, 256, 0, stream>>>(deg, NN);
    k_count<<<(NE + 255) / 256, 256, 0, stream>>>(edst, deg, NE);
    k_dinv<<<(NN + 255) / 256, 256, 0, stream>>>(deg, dinv, NN);

    // GCN layer 1: [N,512]@[512,256]
    prep(W1, 512, 256, 256, 1);
    gemm(nodes, bufA, nullptr, 512, 256, 256, 0);
    gcn_agg(bufA, b1, bufB, 256);

    // GCN layer 2: [N,256]@[256,192]
    prep(W2, 256, 192, 192, 1);
    gemm(bufB, bufA, nullptr, 256, 192, 192, 0);
    gcn_agg(bufA, b2, bufB, 192);

    // GCN layer 3: [N,192]@[192,128]
    prep(W3, 192, 128, 128, 1);
    gemm(bufB, bufA, nullptr, 192, 128, 128, 0);
    gcn_agg(bufA, b3, bufB, 128);

    // fc1 (+relu), fc2
    prep(fc1w, 128, 128, 128, 1);
    gemm(bufB, bufA, fc1b, 128, 128, 128, 1);
    prep(fc2w, 128, 128, 128, 1);
    gemm(bufA, bufB, fc2b, 128, 128, 128, 0);

    // classifier: h @ label^T ; label is [1000,128] == Bt layout directly
    prep(label, 128, 1000, 1024, 0);
    gemm(bufB, out, nullptr, 128, 1000, 1024, 0);
}
